// MLP_89687507075104
// MI455X (gfx1250) — matrix operand order repaired and checked by reading, not compiled, not run on hardware
//
#include <hip/hip_runtime.h>

typedef __attribute__((ext_vector_type(16))) _Float16 v16h;
typedef __attribute__((ext_vector_type(8)))  float    v8f;

#define DEPTH 64
#define TPB   256

// A[m,k] for one layer: W as two diagonal 5x5 blocks plus a bias column:
//   block0: W at (M 0..4,  K 0..4),   bias at (M 0..4,  K=5)
//   block1: W at (M 8..12, K 16..20), bias at (M 8..12, K=21)
// B carries a constant 1.0 at K=5 / K=21, so D = W*h + b with C = inline 0.
__device__ __forceinline__ float fetch_a(const float* __restrict__ Ws,
                                         const float* __restrict__ bs,
                                         int l, int m, int k) {
    if (m < 5) {
        if (k < 5)  return Ws[l * 25 + m * 5 + k];
        if (k == 5) return bs[l * 5 + m];
        return 0.0f;
    }
    if (m >= 8 && m < 13) {
        if (k >= 16 && k < 21) return Ws[l * 25 + (m - 8) * 5 + (k - 16)];
        if (k == 21)           return bs[l * 5 + (m - 8)];
    }
    return 0.0f;
}

// Single-instruction ReLU; fmaxf/fmed3 get expanded to canonicalize+max by the
// backend, so pin it with asm. max_num(0, NaN) = 0 matches fmaxf semantics.
__device__ __forceinline__ float relu1(float v) {
    float r;
    asm("v_max_num_f32 %0, 0, %1" : "=v"(r) : "v"(v));
    return r;
}

__global__ __launch_bounds__(TPB) void mlp64_wmma_kernel(
    const float* __restrict__ x,   // [N,5]
    const float* __restrict__ Ws,  // [64,5,5]
    const float* __restrict__ bs,  // [64,5]
    const float* __restrict__ Wo,  // [1,5]
    const float* __restrict__ bo,  // [1]
    float* __restrict__ out,       // [N]
    int N)
{
    // Weights+bias pre-packed into the exact WMMA A-register image (f16 pairs):
    // 64 layers x 32 lanes x 8 dwords = 64 KB LDS.
    __shared__ unsigned int ldsA[DEPTH * 256];

    {
        const int t  = threadIdx.x;   // 256 threads = 32 lanes x 8 dwords
        const int ln = t >> 3;        // wmma lane 0..31
        const int dw = t & 7;         // dword within lane
        const int m  = ln & 15;       // A row this lane holds
        // 16-bit A 16x32 layout (contiguous): lanes 0-15 hold K 0..15 (v0-7);
        // lanes 16-31 hold K 16..31.
        int kb;
        if (ln < 16) kb = dw * 2;
        else         kb = 16 + dw * 2;
        for (int l = 0; l < DEPTH; ++l) {
            union { _Float16 h[2]; unsigned int u; } p;
            p.h[0] = (_Float16)fetch_a(Ws, bs, l, m, kb);
            p.h[1] = (_Float16)fetch_a(Ws, bs, l, m, kb + 1);
            ldsA[l * 256 + ln * 8 + dw] = p.u;
        }
    }
    __syncthreads();

    const float wo0 = Wo[0], wo1 = Wo[1], wo2 = Wo[2], wo3 = Wo[3], wo4 = Wo[4];
    const float bout = bo[0];

    const unsigned lane  = threadIdx.x & 31u;
    const unsigned wid   = blockIdx.x * (TPB / 32) + (threadIdx.x >> 5);
    const unsigned nwave = gridDim.x * (TPB / 32);

    // Two independent 32-sample streams per wave (64 samples/chunk): the two
    // WMMAs share one A load, and each stream's ReLU/cvt fills the other's
    // WMMA->VALU hazard shadow. Lanes 0-15 = K 0..5 / rows 0..4; lanes 16-31 =
    // K 16..21 / rows 8..12. Stream E: sample base+lane; stream O: base+32+lane.
    for (unsigned base = wid * 64u; base < (unsigned)N; base += nwave * 64u) {
        const unsigned nbase = base + nwave * 64u;
        if (nbase < (unsigned)N) {  // uniform branch: prefetch next chunk's x
            __builtin_prefetch(x + (size_t)(nbase + lane) * 5u, 0, 1);
            __builtin_prefetch(x + (size_t)(nbase + 32u + lane) * 5u, 0, 1);
        }

        const unsigned sE = base + lane;
        const unsigned sO = base + 32u + lane;
        const unsigned cE_idx = (sE < (unsigned)N) ? sE : (unsigned)(N - 1);
        const unsigned cO_idx = (sO < (unsigned)N) ? sO : (unsigned)(N - 1);
        const float* xe = x + (size_t)cE_idx * 5u;
        const float* xo = x + (size_t)cO_idx * 5u;

        float e0 = xe[0], e1 = xe[1], e2 = xe[2], e3 = xe[3], e4 = xe[4];
        float o0 = xo[0], o1 = xo[1], o2 = xo[2], o3 = xo[3], o4 = xo[4];

        v16h bE = {}, bO = {};  // K slots 6..15 / 22..31 stay zero
        bE[0] = (_Float16)e0; bE[1] = (_Float16)e1; bE[2] = (_Float16)e2;
        bE[3] = (_Float16)e3; bE[4] = (_Float16)e4; bE[5] = (_Float16)1.0f;
        bO[0] = (_Float16)o0; bO[1] = (_Float16)o1; bO[2] = (_Float16)o2;
        bO[3] = (_Float16)o3; bO[4] = (_Float16)o4; bO[5] = (_Float16)1.0f;

        #pragma unroll 4
        for (int l = 0; l < DEPTH; ++l) {
            const v16h a = *(const v16h*)&ldsA[l * 256 + lane * 8];
            v8f cE = {}, cO = {};  // inline SRC2 = 0 (bias rides in A/B)
            cE = __builtin_amdgcn_wmma_f32_16x16x32_f16(
                    false, a, false, bE, (short)0, cE, false, false);
            cO = __builtin_amdgcn_wmma_f32_16x16x32_f16(
                    false, a, false, bO, (short)0, cO, false, false);
            e0 = relu1(cE[0]); e1 = relu1(cE[1]); e2 = relu1(cE[2]);
            e3 = relu1(cE[3]); e4 = relu1(cE[4]);
            o0 = relu1(cO[0]); o1 = relu1(cO[1]); o2 = relu1(cO[2]);
            o3 = relu1(cO[3]); o4 = relu1(cO[4]);
            bE[0] = (_Float16)e0; bE[1] = (_Float16)e1; bE[2] = (_Float16)e2;
            bE[3] = (_Float16)e3; bE[4] = (_Float16)e4;
            bO[0] = (_Float16)o0; bO[1] = (_Float16)o1; bO[2] = (_Float16)o2;
            bO[3] = (_Float16)o3; bO[4] = (_Float16)o4;
        }

        const float rE = fmaf(e0, wo0, fmaf(e1, wo1, fmaf(e2, wo2,
                         fmaf(e3, wo3, fmaf(e4, wo4, bout)))));
        const float rO = fmaf(o0, wo0, fmaf(o1, wo1, fmaf(o2, wo2,
                         fmaf(o3, wo3, fmaf(o4, wo4, bout)))));
        if (sE < (unsigned)N) out[sE] = rE;
        if (sO < (unsigned)N) out[sO] = rO;
    }
}

extern "C" void kernel_launch(void* const* d_in, const int* in_sizes, int n_in,
                              void* d_out, int out_size, void* d_ws, size_t ws_size,
                              hipStream_t stream) {
    const float* x  = (const float*)d_in[0];
    const float* Ws = (const float*)d_in[1];
    const float* bs = (const float*)d_in[2];
    const float* Wo = (const float*)d_in[3];
    const float* bo = (const float*)d_in[4];
    float* out = (float*)d_out;

    const int N = in_sizes[0] / 5;
    int waves  = (N + 63) / 64;   // 64 samples per wave-chunk (2 streams)
    int blocks = (waves + (TPB / 32) - 1) / (TPB / 32);
    if (blocks > 1024) blocks = 1024;  // grid-stride: 64KB LDS/WG, ~5 WGs/WGP
    if (blocks < 1)    blocks = 1;

    hipLaunchKernelGGL(mlp64_wmma_kernel, dim3(blocks), dim3(TPB), 0, stream,
                       x, Ws, bs, Wo, bo, out, N);
}
